// GNNModel_60155311948230
// MI455X (gfx1250) — matrix operand order repaired and checked by reading, not compiled, not run on hardware
//
#include <hip/hip_runtime.h>

// ---------------------------------------------------------------------------
// GNN (3-class, 3-layer GCN) for MI455X / gfx1250, wave32.
// GEMMs use v_wmma_f32_16x16x32_bf16 with hi/lo bf16 split (~fp32 accuracy).
// bf16 splitting is done once in producer kernels; the WMMA loop is pure
// b128 loads + wmma. Edge aggregation is L2-resident float4 gather +
// global fp32 atomics.
// ---------------------------------------------------------------------------

#define CDIV(a, b) (((a) + (b) - 1) / (b))

static constexpr int Dm = 128;  // feature dim
static constexpr int Cc = 3;    // classes
static constexpr int Ll = 3;    // mp layers
static constexpr int Gg = 2048; // graphs

typedef __attribute__((ext_vector_type(16))) __bf16 v16bf;
typedef __attribute__((ext_vector_type(8)))  __bf16 v8bf;
typedef __attribute__((ext_vector_type(4)))  __bf16 v4bf;
typedef __attribute__((ext_vector_type(8)))  float  v8f;

__device__ __forceinline__ __bf16 f2bf(float f) {
    unsigned int u = __builtin_bit_cast(unsigned int, f);
    unsigned int r = (u + 0x7FFFu + ((u >> 16) & 1u)) >> 16;  // RNE
    unsigned short s = (unsigned short)r;
    return __builtin_bit_cast(__bf16, s);
}
__device__ __forceinline__ float bf2f(__bf16 b) {
    unsigned short s = __builtin_bit_cast(unsigned short, b);
    unsigned int u = ((unsigned int)s) << 16;
    return __builtin_bit_cast(float, u);
}
struct bfpair { __bf16 hi, lo; };
__device__ __forceinline__ bfpair split_hilo(float v) {
    bfpair p;
    p.hi = f2bf(v);
    p.lo = f2bf(v - bf2f(p.hi));
    return p;
}

// ---------------------------------------------------------------------------
// Elementwise / setup kernels
// ---------------------------------------------------------------------------
__global__ void fill_kernel(float* __restrict__ p, float v, long n) {
    long i = (long)blockIdx.x * blockDim.x + threadIdx.x;
    if (i < n) p[i] = v;
}

__global__ void deg_count_kernel(const int* __restrict__ col, float* __restrict__ deg, int E) {
    long e = (long)blockIdx.x * blockDim.x + threadIdx.x;
    if (e < E) atomicAdd(&deg[col[e]], 1.0f);
}

// deg (in) -> deg := 1/deg (self-loop coeff), dinv := rsqrt(deg)
__global__ void deg_fin_kernel(float* __restrict__ deg, float* __restrict__ dinv, int N) {
    long i = (long)blockIdx.x * blockDim.x + threadIdx.x;
    if (i >= N) return;
    float d = deg[i];
    dinv[i] = rsqrtf(d);
    deg[i]  = 1.0f / d;
}

__global__ void enorm_kernel(const int* __restrict__ row, const int* __restrict__ col,
                             const float* __restrict__ dinv, float* __restrict__ en, int E) {
    long e = (long)blockIdx.x * blockDim.x + threadIdx.x;
    if (e < E) en[e] = dinv[row[e]] * dinv[col[e]];
}

// Generic fp32 -> bf16 hi/lo split (row-major, elementwise).
__global__ void split_kernel(const float* __restrict__ src, __bf16* __restrict__ hi,
                             __bf16* __restrict__ lo, long n) {
    long i = (long)blockIdx.x * blockDim.x + threadIdx.x;
    if (i >= n) return;
    bfpair p = split_hilo(src[i]);
    hi[i] = p.hi;
    lo[i] = p.lo;
}

// Transpose + hi/lo bf16 split of the 12 [128x128] weight mats (9 conv + 3 lin1).
// Source layout [k][n] (h @ W), dest layout [n][k] so B-fragments are contiguous in K.
__global__ void wprep_kernel(const float* __restrict__ conv_w, const float* __restrict__ lin1_w,
                             __bf16* __restrict__ whi, __bf16* __restrict__ wlo) {
    long idx = (long)blockIdx.x * blockDim.x + threadIdx.x;
    if (idx >= (long)(Cc * Ll + Cc) * Dm * Dm) return;
    long mat = idx >> 14;            // / (128*128)
    int  rem = (int)(idx & 16383);
    int  n = rem >> 7;
    int  k = rem & 127;
    float v = (mat < Cc * Ll) ? conv_w[(mat << 14) + (long)k * Dm + n]
                              : lin1_w[((mat - Cc * Ll) << 14) + (long)k * Dm + n];
    bfpair p = split_hilo(v);
    whi[(mat << 14) + (long)n * Dm + k] = p.hi;
    wlo[(mat << 14) + (long)n * Dm + k] = p.lo;
}

// ---------------------------------------------------------------------------
// WMMA GEMM: T[M x 128] = A[M x 128] @ W, both as bf16 hi/lo pairs.
// A row-major [m][k]; W transposed [n][k]. One 16x16 output tile per wave32,
// 8 waves/block = one 16-row x 128-col strip. Grid = M/16 blocks.
// Optional fused bias + relu epilogue.
// ---------------------------------------------------------------------------
__global__ __launch_bounds__(256) void gemm_wmma_kernel(
    const __bf16* __restrict__ Ahi, const __bf16* __restrict__ Alo,
    const __bf16* __restrict__ Wt_hi, const __bf16* __restrict__ Wt_lo,
    float* __restrict__ T, const float* __restrict__ bias, int relu) {
    const int wave = threadIdx.x >> 5;    // n-tile 0..7
    const int lane = threadIdx.x & 31;
    const int tm   = blockIdx.x;          // m-tile
    const int half = lane >> 4;           // 0: lanes 0-15, 1: lanes 16-31
    const int lid  = lane & 15;
    const int row0 = tm * 16;
    const int n0   = wave * 16;

    v8f acc = {};
#pragma unroll
    for (int kb = 0; kb < Dm; kb += 32) {
        // A fragment (16-bit A 16x32 layout): row m = row0+lid;
        // lanes<16: K runs kb+[0..7], kb+16+[0..7]; lanes>=16: both +8.
        const long abase = (long)(row0 + lid) * Dm + kb + half * 8;
        v8bf ah0 = *(const v8bf*)(Ahi + abase);
        v8bf ah1 = *(const v8bf*)(Ahi + abase + 16);
        v8bf al0 = *(const v8bf*)(Alo + abase);
        v8bf al1 = *(const v8bf*)(Alo + abase + 16);
        v16bf a_hi = __builtin_shufflevector(ah0, ah1, 0, 1, 2, 3, 4, 5, 6, 7,
                                             8, 9, 10, 11, 12, 13, 14, 15);
        v16bf a_lo = __builtin_shufflevector(al0, al1, 0, 1, 2, 3, 4, 5, 6, 7,
                                             8, 9, 10, 11, 12, 13, 14, 15);
        // B fragment (16-bit B 32x16 layout): col n = n0+lid;
        // lanes<16: K runs kb+[0..7], kb+16+[0..7]; lanes>=16: both +8 (same as A).
        const long wofs = (long)(n0 + lid) * Dm + kb + half * 8;
        v8bf bh0 = *(const v8bf*)(Wt_hi + wofs);
        v8bf bh1 = *(const v8bf*)(Wt_hi + wofs + 16);
        v8bf bl0 = *(const v8bf*)(Wt_lo + wofs);
        v8bf bl1 = *(const v8bf*)(Wt_lo + wofs + 16);
        v16bf b_hi = __builtin_shufflevector(bh0, bh1, 0, 1, 2, 3, 4, 5, 6, 7,
                                             8, 9, 10, 11, 12, 13, 14, 15);
        v16bf b_lo = __builtin_shufflevector(bl0, bl1, 0, 1, 2, 3, 4, 5, 6, 7,
                                             8, 9, 10, 11, 12, 13, 14, 15);

        // hi*hi + hi*lo + lo*hi ~= fp32 product, accumulated in fp32.
        acc = __builtin_amdgcn_wmma_f32_16x16x32_bf16(false, a_hi, false, b_hi,
                                                      (short)0, acc, false, false);
        acc = __builtin_amdgcn_wmma_f32_16x16x32_bf16(false, a_hi, false, b_lo,
                                                      (short)0, acc, false, false);
        acc = __builtin_amdgcn_wmma_f32_16x16x32_bf16(false, a_lo, false, b_hi,
                                                      (short)0, acc, false, false);
    }
    // C/D layout: lane (half,lid) holds (M = i + 8*half, N = lid), i = VGPR 0..7.
    const int n = n0 + lid;
    float bv = bias ? bias[n] : 0.0f;
#pragma unroll
    for (int i = 0; i < 8; ++i) {
        float v = acc[i] + bv;
        if (relu) v = fmaxf(v, 0.0f);
        T[(long)(row0 + half * 8 + i) * Dm + n] = v;
    }
}

// agg = t * sinv[row] + bias[d]   (self-loop term + conv bias)
__global__ void agg_init_kernel(const float* __restrict__ T, const float* __restrict__ sinv,
                                const float* __restrict__ bias, float* __restrict__ agg, int N) {
    long idx = (long)blockIdx.x * blockDim.x + threadIdx.x;
    if (idx >= (long)N * Dm) return;
    long i = idx >> 7;
    int  d = (int)(idx & 127);
    agg[idx] = T[idx] * sinv[i] + bias[d];
}

// agg[col] += t[row] * enorm : one wave32 per edge, float4 per lane.
__global__ __launch_bounds__(256) void edge_agg_kernel(
    const float* __restrict__ T, const int* __restrict__ row, const int* __restrict__ col,
    const float* __restrict__ en, float* __restrict__ agg, int E) {
    long idx = (long)blockIdx.x * blockDim.x + threadIdx.x;
    long e = idx >> 5;
    int lane = (int)(idx & 31);
    if (e >= E) return;
    int r = row[e], c = col[e];
    float w = en[e];
    const float4 v = *(const float4*)(T + (long)r * Dm + lane * 4);
    float* dst = agg + (long)c * Dm + lane * 4;
    atomicAdd(dst + 0, v.x * w);
    atomicAdd(dst + 1, v.y * w);
    atomicAdd(dst + 2, v.z * w);
    atomicAdd(dst + 3, v.w * w);
}

// h = relu(F.normalize(agg, dim=1)); also emits bf16 hi/lo for the next GEMM.
// One wave32 per row, shuffle reduction (wave32-native).
__global__ __launch_bounds__(256) void l2norm_relu_kernel(
    const float* __restrict__ A, float* __restrict__ H,
    __bf16* __restrict__ Hhi, __bf16* __restrict__ Hlo, int N) {
    long idx = (long)blockIdx.x * blockDim.x + threadIdx.x;
    long rrow = idx >> 5;
    int lane = (int)(idx & 31);
    if (rrow >= N) return;
    const float4 v = *(const float4*)(A + rrow * Dm + lane * 4);
    float s = v.x * v.x + v.y * v.y + v.z * v.z + v.w * v.w;
#pragma unroll
    for (int o = 16; o > 0; o >>= 1) s += __shfl_xor(s, o, 32);
    float inv = 1.0f / fmaxf(sqrtf(s), 1e-12f);
    float4 o4;
    o4.x = fmaxf(v.x * inv, 0.0f);
    o4.y = fmaxf(v.y * inv, 0.0f);
    o4.z = fmaxf(v.z * inv, 0.0f);
    o4.w = fmaxf(v.w * inv, 0.0f);
    *(float4*)(H + rrow * Dm + lane * 4) = o4;
    bfpair p0 = split_hilo(o4.x);
    bfpair p1 = split_hilo(o4.y);
    bfpair p2 = split_hilo(o4.z);
    bfpair p3 = split_hilo(o4.w);
    v4bf hi4 = {p0.hi, p1.hi, p2.hi, p3.hi};
    v4bf lo4 = {p0.lo, p1.lo, p2.lo, p3.lo};
    // pack 4 bf16 (8 bytes) per lane
    *(v4bf*)(Hhi + rrow * Dm + lane * 4) = hi4;
    *(v4bf*)(Hlo + rrow * Dm + lane * 4) = lo4;
}

// pooled[batch[i]] += h[i] : one wave32 per node.
__global__ __launch_bounds__(256) void pool_kernel(
    const float* __restrict__ H, const int* __restrict__ batch,
    float* __restrict__ pooled, int N) {
    long idx = (long)blockIdx.x * blockDim.x + threadIdx.x;
    long i = idx >> 5;
    int lane = (int)(idx & 31);
    if (i >= N) return;
    int g = batch[i];
    const float4 v = *(const float4*)(H + i * Dm + lane * 4);
    float* dst = pooled + (long)g * Dm + lane * 4;
    atomicAdd(dst + 0, v.x);
    atomicAdd(dst + 1, v.y);
    atomicAdd(dst + 2, v.z);
    atomicAdd(dst + 3, v.w);
}

// out[g, c] = dot(z[g], lin2_w[c]) + lin2_b[c] : one wave32 per graph.
__global__ __launch_bounds__(256) void lin2_kernel(
    const float* __restrict__ Z, const float* __restrict__ W2,
    const float* __restrict__ B2, float* __restrict__ out, int c) {
    long idx = (long)blockIdx.x * blockDim.x + threadIdx.x;
    long g = idx >> 5;
    int lane = (int)(idx & 31);
    if (g >= Gg) return;
    const float4 z4 = *(const float4*)(Z + g * Dm + lane * 4);
    const float4 w4 = *(const float4*)(W2 + (long)c * Dm + lane * 4);
    float s = z4.x * w4.x + z4.y * w4.y + z4.z * w4.z + z4.w * w4.w;
#pragma unroll
    for (int o = 16; o > 0; o >>= 1) s += __shfl_xor(s, o, 32);
    if (lane == 0) out[g * Cc + c] = s + B2[c];
}

// ---------------------------------------------------------------------------
extern "C" void kernel_launch(void* const* d_in, const int* in_sizes, int n_in,
                              void* d_out, int out_size, void* d_ws, size_t ws_size,
                              hipStream_t stream) {
    const float* x      = (const float*)d_in[0];
    const int*   eidx   = (const int*)d_in[1];
    const int*   batch  = (const int*)d_in[2];
    const float* conv_w = (const float*)d_in[3];
    const float* conv_b = (const float*)d_in[4];
    const float* lin1_w = (const float*)d_in[5];
    const float* lin1_b = (const float*)d_in[6];
    const float* lin2_w = (const float*)d_in[7];
    const float* lin2_b = (const float*)d_in[8];
    float* out = (float*)d_out;

    const int N = in_sizes[2];       // 50000 (multiple of 16)
    const int E = in_sizes[1] / 2;   // 1.6M
    const int* row = eidx;
    const int* col = eidx + E;

    // --- carve workspace (256B-aligned sub-allocations) ---
    size_t off = 0;
    auto carve = [&](size_t bytes) -> char* {
        char* p = (char*)d_ws + off;
        off = (off + bytes + 255) & ~(size_t)255;
        return p;
    };
    float*  deg    = (float*)carve((size_t)N * 4);            // becomes sinv = 1/deg
    float*  dinv   = (float*)carve((size_t)N * 4);
    float*  en     = (float*)carve((size_t)E * 4);
    float*  h      = (float*)carve((size_t)N * Dm * 4);
    float*  t      = (float*)carve((size_t)N * Dm * 4);
    float*  agg    = (float*)carve((size_t)N * Dm * 4);
    float*  pooled = (float*)carve((size_t)Gg * Dm * 4);
    float*  z      = (float*)carve((size_t)Gg * Dm * 4);
    __bf16* whi    = (__bf16*)carve((size_t)(Cc * Ll + Cc) * Dm * Dm * 2);
    __bf16* wlo    = (__bf16*)carve((size_t)(Cc * Ll + Cc) * Dm * Dm * 2);
    __bf16* xhi    = (__bf16*)carve((size_t)N * Dm * 2);
    __bf16* xlo    = (__bf16*)carve((size_t)N * Dm * 2);
    __bf16* hhi    = (__bf16*)carve((size_t)N * Dm * 2);
    __bf16* hlo    = (__bf16*)carve((size_t)N * Dm * 2);
    __bf16* phi    = (__bf16*)carve((size_t)Gg * Dm * 2);
    __bf16* plo    = (__bf16*)carve((size_t)Gg * Dm * 2);

    // --- degrees, edge coefficients, weight/x prep (reused by all passes) ---
    fill_kernel<<<CDIV((long)N, 256), 256, 0, stream>>>(deg, 1.0f, N);
    deg_count_kernel<<<CDIV((long)E, 256), 256, 0, stream>>>(col, deg, E);
    deg_fin_kernel<<<CDIV((long)N, 256), 256, 0, stream>>>(deg, dinv, N);
    enorm_kernel<<<CDIV((long)E, 256), 256, 0, stream>>>(row, col, dinv, en, E);
    wprep_kernel<<<CDIV((long)(Cc * Ll + Cc) * Dm * Dm, 256), 256, 0, stream>>>(
        conv_w, lin1_w, whi, wlo);
    split_kernel<<<CDIV((long)N * Dm, 256), 256, 0, stream>>>(x, xhi, xlo, (long)N * Dm);

    for (int c = 0; c < Cc; ++c) {
        for (int l = 0; l < Ll; ++l) {
            const int mat = c * Ll + l;
            const __bf16* Ahi = (l == 0) ? xhi : hhi;
            const __bf16* Alo = (l == 0) ? xlo : hlo;
            gemm_wmma_kernel<<<N / 16, 256, 0, stream>>>(
                Ahi, Alo, whi + (size_t)mat * Dm * Dm, wlo + (size_t)mat * Dm * Dm,
                t, nullptr, 0);
            agg_init_kernel<<<CDIV((long)N * Dm, 256), 256, 0, stream>>>(
                t, deg, conv_b + (size_t)mat * Dm, agg, N);
            edge_agg_kernel<<<CDIV((long)E * 32, 256), 256, 0, stream>>>(
                t, row, col, en, agg, E);
            l2norm_relu_kernel<<<CDIV((long)N * 32, 256), 256, 0, stream>>>(
                agg, h, hhi, hlo, N);
        }
        fill_kernel<<<CDIV((long)Gg * Dm, 256), 256, 0, stream>>>(pooled, 0.0f, (long)Gg * Dm);
        pool_kernel<<<CDIV((long)N * 32, 256), 256, 0, stream>>>(h, batch, pooled, N);
        split_kernel<<<CDIV((long)Gg * Dm, 256), 256, 0, stream>>>(
            pooled, phi, plo, (long)Gg * Dm);
        const int mat = Cc * Ll + c;
        gemm_wmma_kernel<<<Gg / 16, 256, 0, stream>>>(
            phi, plo, whi + (size_t)mat * Dm * Dm, wlo + (size_t)mat * Dm * Dm,
            z, lin1_b + (size_t)c * Dm, 1);
        lin2_kernel<<<CDIV((long)Gg * 32, 256), 256, 0, stream>>>(z, lin2_w, lin2_b, out, c);
    }
}
